// AttentionTemporelle_83915071029420
// MI455X (gfx1250) — compile-verified, matrix operand order checked by reading, not run on hardware
//
#include <hip/hip_runtime.h>

#define B_   8
#define T_   2048
#define D_   512
#define DK_  128
#define NKT  47          // key tiles covering the +/-360 band around a 16-row query tile
#define SW   768         // padded score strip width (48 tiles * 16)

__device__ __forceinline__ int imin(int a, int b){ return a < b ? a : b; }
__device__ __forceinline__ int imax(int a, int b){ return a > b ? a : b; }
__device__ __forceinline__ int iabs(int a){ return a < 0 ? -a : a; }

typedef __attribute__((ext_vector_type(16))) __bf16 v16bf;
typedef __attribute__((ext_vector_type(8)))  float  v8f;

union Frag { unsigned int u[8]; v16bf v; };

__device__ __forceinline__ unsigned short f2bf(float f){
    unsigned int u = __builtin_bit_cast(unsigned int, f);
    u += 0x7fffu + ((u >> 16) & 1u);        // round to nearest even
    return (unsigned short)(u >> 16);
}

__device__ __forceinline__ v8f wmma_bf16(v16bf a, v16bf b, v8f c){
    return __builtin_amdgcn_wmma_f32_16x16x32_bf16(false, a, false, b, (short)0, c, false, false);
}

// ---- fragment loaders (ISA 7.12.2 16-bit layouts) -------------------------
// A (16x32 MxK): element (m,k) at s[m*stride + k]; K-pairs contiguous
__device__ __forceinline__ v16bf load_a_lds(const unsigned short* s, int stride, int lane){
    int half = (lane >> 4) & 1, m = lane & 15;
    Frag f;
#pragma unroll
    for(int v = 0; v < 8; ++v){
        int kb = ((v < 4) ? 0 : 16) + (v & 3) * 2 + half * 8;
        f.u[v] = *(const unsigned int*)(s + m * stride + kb);
    }
    return f.v;
}

// B (32x16 KxN) with K contiguous: element (k,n) at base[n*nstride + k]
__device__ __forceinline__ v16bf load_b_kcontig(const unsigned short* base, size_t nstride, int lane){
    int half = (lane >> 4) & 1, n = lane & 15;
    const unsigned short* p = base + (size_t)n * nstride;
    Frag f;
#pragma unroll
    for(int v = 0; v < 8; ++v){
        int kb = ((v < 4) ? 0 : 16) + (v & 3) * 2 + half * 8;
        f.u[v] = *(const unsigned int*)(p + kb);
    }
    return f.v;
}

// B = k^T (32x16 KxN): B[k][n] = K[j0+n][kk+k]; K-pairs contiguous in a key row
__device__ __forceinline__ v16bf load_b_ktrans(const unsigned short* kbase, int j0, int kk, int lane){
    int half = (lane >> 4) & 1, n = lane & 15;
    int j = imin(imax(j0 + n, 0), T_ - 1);          // clamp; invalid cols masked later
    const unsigned short* row = kbase + (size_t)j * DK_ + kk;
    Frag f;
#pragma unroll
    for(int v = 0; v < 8; ++v){
        int kb = ((v < 4) ? 0 : 16) + (v & 3) * 2 + half * 8;
        f.u[v] = *(const unsigned int*)(row + kb);
    }
    return f.v;
}

// B (32x16 KxN) from transposed x: B[k][n] = xT[col0+n][j0+k]
// Wave-uniform fast path keeps offsets contiguous so loads merge to b128.
__device__ __forceinline__ v16bf load_b_xT(const unsigned short* xTb, int j0, int col0, int lane){
    int half = (lane >> 4) & 1, n = lane & 15;
    const unsigned short* p = xTb + (size_t)(col0 + n) * T_;
    Frag f;
    if (j0 >= 0 && j0 <= T_ - 32){                  // whole 32-row span valid (uniform)
#pragma unroll
        for(int v = 0; v < 8; ++v){
            int kb = ((v < 4) ? 0 : 16) + (v & 3) * 2 + half * 8;
            f.u[v] = *(const unsigned int*)(p + j0 + kb);
        }
    } else {
#pragma unroll
        for(int v = 0; v < 8; ++v){
            int kb = ((v < 4) ? 0 : 16) + (v & 3) * 2 + half * 8;
            int j = imin(imax(j0 + kb, 0), T_ - 2); // garbage x 0-weight = 0
            f.u[v] = *(const unsigned int*)(p + j);
        }
    }
    return f.v;
}

// ---------------------------------------------------------------------------
// Tiled transpose + f32->bf16: out[c][r] = bf16(in[r][c]), R,C multiples of 32
__global__ __launch_bounds__(256)
void k_wT(const float* __restrict__ in, unsigned short* __restrict__ out, int R, int C){
    __shared__ float tile[32][33];
    int r0 = blockIdx.x * 32, c0 = blockIdx.y * 32;
    for (int i = threadIdx.y; i < 32; i += 8)
        tile[i][threadIdx.x] = in[(size_t)(r0 + i) * C + c0 + threadIdx.x];
    __syncthreads();
    for (int i = threadIdx.y; i < 32; i += 8)
        out[(size_t)(c0 + i) * R + r0 + threadIdx.x] = f2bf(tile[threadIdx.x][i]);
}

// xT[b][c][t] = bf16(x[b][t][c])
__global__ __launch_bounds__(256)
void k_xT(const float* __restrict__ x, unsigned short* __restrict__ xT){
    __shared__ float tile[32][33];
    int b = blockIdx.z;
    int t0 = blockIdx.x * 32, c0 = blockIdx.y * 32;
    const float* xb = x + (size_t)b * T_ * D_;
    unsigned short* xTb = xT + (size_t)b * D_ * T_;
    for (int i = threadIdx.y; i < 32; i += 8)
        tile[i][threadIdx.x] = xb[(size_t)(t0 + i) * D_ + c0 + threadIdx.x];
    __syncthreads();
    for (int i = threadIdx.y; i < 32; i += 8)
        xTb[(size_t)(c0 + i) * T_ + t0 + threadIdx.x] = f2bf(tile[threadIdx.x][i]);
}

// q = x@Wq + bq, k = x@Wk + bk  (one 16-row tile per WG, 8 waves = 8 col tiles)
__global__ __launch_bounds__(256)
void k_proj(const float* __restrict__ x,
            const unsigned short* __restrict__ WqT, const float* __restrict__ bq,
            const unsigned short* __restrict__ WkT, const float* __restrict__ bk,
            unsigned short* __restrict__ qbf, unsigned short* __restrict__ kbf){
    __shared__ unsigned short xs[16 * D_];
    int row0 = blockIdx.x * 16;
    int tid = threadIdx.x, lane = tid & 31, wave = tid >> 5;
    // stage x tile f32 -> bf16 (packed uint writes)
    const float2* xrow = (const float2*)(x + (size_t)row0 * D_);
    unsigned int* xsp = (unsigned int*)xs;
    for (int i = tid; i < 16 * D_ / 2; i += 256){
        float2 v2 = xrow[i];
        xsp[i] = (unsigned int)f2bf(v2.x) | ((unsigned int)f2bf(v2.y) << 16);
    }
    __syncthreads();
    v8f accq = {}, acck = {};
    int ncol0 = wave * 16;                       // DK=128 -> 8 column tiles
    int n = lane & 15;
#pragma unroll 2
    for (int kk = 0; kk < D_; kk += 32){
        __builtin_prefetch(WqT + (size_t)(ncol0 + n) * D_ + kk + 64, 0, 1);
        __builtin_prefetch(WkT + (size_t)(ncol0 + n) * D_ + kk + 64, 0, 1);
        v16bf a   = load_a_lds(xs + kk, D_, lane);
        v16bf bqf = load_b_kcontig(WqT + (size_t)ncol0 * D_ + kk, D_, lane);
        v16bf bkf = load_b_kcontig(WkT + (size_t)ncol0 * D_ + kk, D_, lane);
        accq = wmma_bf16(a, bqf, accq);
        acck = wmma_bf16(a, bkf, acck);
    }
    int half = lane >> 4, col = ncol0 + n;
    float biasq = bq[col], biask = bk[col];
#pragma unroll
    for (int r = 0; r < 8; ++r){
        int m = r + half * 8;
        qbf[(size_t)(row0 + m) * DK_ + col] = f2bf(accq[r] + biasq);
        kbf[(size_t)(row0 + m) * DK_ + col] = f2bf(acck[r] + biask);
    }
}

// Banded attention: acc = (1/3) * sum_w softmax_w(q k^T / sqrt(dk)) @ x
__global__ __launch_bounds__(256)
void k_attn(const unsigned short* __restrict__ qbf, const unsigned short* __restrict__ kbf,
            const unsigned short* __restrict__ xT, unsigned short* __restrict__ accbf){
    __shared__ unsigned short qs[16 * DK_];     //  4 KB
    __shared__ float          sc[16 * SW];      // 48 KB banded score strip
    __shared__ unsigned short pw[16 * SW];      // 24 KB combined weights (bf16)
    __shared__ float mrow[3][16], invl[3][16];
    __shared__ float pmax[16][16], psum[16][16];

    int tile = blockIdx.x;
    int b  = tile / (T_ / 16);
    int t0 = (tile % (T_ / 16)) * 16;
    int jlo = t0 - 368;                         // tile-aligned band start (may be < 0)
    int tid = threadIdx.x, lane = tid & 31, wave = tid >> 5;
    const unsigned short* kbase = kbf + (size_t)b * T_ * DK_;
    const unsigned short* xTb   = xT  + (size_t)b * D_ * T_;
    const float scale = 0.0883883476483184f;    // 1/sqrt(128)

    {
        const unsigned int* qsrc = (const unsigned int*)(qbf + (size_t)(b * T_ + t0) * DK_);
        unsigned int* qdst = (unsigned int*)qs;
        for (int i = tid; i < 16 * DK_ / 2; i += 256) qdst[i] = qsrc[i];
    }
    __syncthreads();

    // Phase 1: scores (16 x 752) via WMMA, waves split the key tiles
    for (int kt = wave; kt < NKT; kt += 8){
        int j0 = jlo + kt * 16;
        v8f c = {};
#pragma unroll
        for (int kk = 0; kk < DK_; kk += 32){
            v16bf a  = load_a_lds(qs + kk, DK_, lane);
            v16bf bb = load_b_ktrans(kbase, j0, kk, lane);
            c = wmma_bf16(a, bb, c);
        }
        int half = lane >> 4, n = lane & 15;
        int j = j0 + n;
        bool valid = (j >= 0) && (j < T_);
#pragma unroll
        for (int r = 0; r < 8; ++r){
            int m = r + half * 8;
            sc[m * SW + kt * 16 + n] = valid ? c[r] * scale : -1e30f;
        }
    }
    __syncthreads();

    // Phase 2a: widest window (hw=360) — 16 chunks per row, online max/sum
    {
        int i = tid >> 4, ch = tid & 15;
        int t = t0 + i;
        int lo = imax(t - 360, 0), hi = imin(t + 360, T_ - 1);
        int len = hi - lo + 1;
        int c0 = lo + (len * ch) / 16, c1 = lo + (len * (ch + 1)) / 16;
        float m = -1e30f, l = 0.f;
        for (int j = c0; j < c1; ++j){
            float s = sc[i * SW + (j - jlo)];
            if (s > m){ l *= __expf(m - s); m = s; }
            l += __expf(s - m);
        }
        pmax[i][ch] = m; psum[i][ch] = l;
    }
    __syncthreads();
    // Phase 2b: combine w2 partials (tid<16); small windows serial (tid 16..47)
    if (tid < 16){
        int i = tid;
        float m = -1e30f;
#pragma unroll
        for (int ch = 0; ch < 16; ++ch) m = fmaxf(m, pmax[i][ch]);
        float l = 0.f;
#pragma unroll
        for (int ch = 0; ch < 16; ++ch) l += psum[i][ch] * __expf(pmax[i][ch] - m);
        mrow[2][i] = m; invl[2][i] = 1.f / l;
    } else if (tid < 48){
        int i = tid & 15, w = (tid >> 4) - 1;   // w = 0 (hw 12) or 1 (hw 84)
        int hw = (w == 0) ? 12 : 84;
        int t = t0 + i;
        int lo = imax(t - hw, 0), hi = imin(t + hw, T_ - 1);
        float m = -1e30f, l = 0.f;
        for (int j = lo; j <= hi; ++j){
            float s = sc[i * SW + (j - jlo)];
            if (s > m){ l *= __expf(m - s); m = s; }
            l += __expf(s - m);
        }
        mrow[w][i] = m; invl[w][i] = 1.f / l;
    }
    __syncthreads();

    // Phase 3: combined bf16 weights P = (1/3) sum_w softmax_w (zero outside bands)
    for (int idx = tid; idx < 16 * SW; idx += 256){
        int i = idx / SW, jrel = idx % SW;
        int j = jlo + jrel;
        float val = 0.f;
        if (j >= 0 && j < T_ && jrel < NKT * 16){
            float s = sc[idx];
            int d = iabs(j - (t0 + i));
            if (d <= 12 ) val += __expf(s - mrow[0][i]) * invl[0][i];
            if (d <= 84 ) val += __expf(s - mrow[1][i]) * invl[1][i];
            if (d <= 360) val += __expf(s - mrow[2][i]) * invl[2][i];
            val *= (1.f / 3.f);
        }
        pw[idx] = f2bf(val);
    }
    __syncthreads();

    // Phase 4: acc(16x512) = P(16x752) @ x_band ; each wave owns 64 columns
    v8f acc[4] = {{}, {}, {}, {}};
    int ncol0 = wave * 64;
    int n15 = lane & 15;
#pragma unroll 2
    for (int ks = 0; ks < SW; ks += 32){
        __builtin_prefetch(xTb + (size_t)(ncol0 + n15) * T_ + imin(imax(jlo + ks + 64, 0), T_ - 1), 0, 1);
        v16bf a = load_a_lds(pw + ks, SW, lane);
#pragma unroll
        for (int nt = 0; nt < 4; ++nt){
            v16bf bb = load_b_xT(xTb, jlo + ks, ncol0 + nt * 16, lane);
            acc[nt] = wmma_bf16(a, bb, acc[nt]);
        }
    }
    int half = lane >> 4;
#pragma unroll
    for (int nt = 0; nt < 4; ++nt){
        int col = ncol0 + nt * 16 + n15;
#pragma unroll
        for (int r = 0; r < 8; ++r){
            int m = r + half * 8;
            accbf[(size_t)(b * T_ + t0 + m) * D_ + col] = f2bf(acc[nt][r]);
        }
    }
}

// out = acc@Wo + bo ; res = x + out ; LayerNorm(res)
__global__ __launch_bounds__(256)
void k_out(const unsigned short* __restrict__ accbf, const unsigned short* __restrict__ WoT,
           const float* __restrict__ bo, const float* __restrict__ x,
           const float* __restrict__ gamma, const float* __restrict__ beta,
           float* __restrict__ out){
    __shared__ unsigned short as_[16 * D_];     // 16 KB
    __shared__ float          res[16 * D_];     // 32 KB
    __shared__ float ps1[16][17], ps2[16][17];
    __shared__ float mu[16], rsig[16];
    int row0 = blockIdx.x * 16;
    int tid = threadIdx.x, lane = tid & 31, wave = tid >> 5;
    {
        const unsigned int* src = (const unsigned int*)(accbf + (size_t)row0 * D_);
        unsigned int* dst = (unsigned int*)as_;
        for (int i = tid; i < 16 * D_ / 2; i += 256) dst[i] = src[i];
    }
    __syncthreads();
    v8f acc[4] = {{}, {}, {}, {}};
    int ncol0 = wave * 64;
    int n15 = lane & 15;
#pragma unroll 2
    for (int kk = 0; kk < D_; kk += 32){
        __builtin_prefetch(WoT + (size_t)(ncol0 + n15) * D_ + kk + 64, 0, 1);
        v16bf a = load_a_lds(as_ + kk, D_, lane);
#pragma unroll
        for (int nt = 0; nt < 4; ++nt){
            v16bf bb = load_b_kcontig(WoT + (size_t)(ncol0 + nt * 16) * D_ + kk, D_, lane);
            acc[nt] = wmma_bf16(a, bb, acc[nt]);
        }
    }
    int half = lane >> 4;
#pragma unroll
    for (int nt = 0; nt < 4; ++nt){
        int col = ncol0 + nt * 16 + n15;
        float bias = bo[col];
#pragma unroll
        for (int r = 0; r < 8; ++r){
            int m = r + half * 8;
            res[m * D_ + col] = x[(size_t)(row0 + m) * D_ + col] + acc[nt][r] + bias;
        }
    }
    __syncthreads();
    // LN stats: 16 chunks per row (sum, sum of squares), then combine
    {
        int i = tid >> 4, ch = tid & 15;
        const float* r = res + i * D_ + ch * 32;
        float s = 0.f, s2 = 0.f;
#pragma unroll
        for (int c = 0; c < 32; ++c){ float v = r[c]; s += v; s2 += v * v; }
        ps1[i][ch] = s; ps2[i][ch] = s2;
    }
    __syncthreads();
    if (tid < 16){
        float s = 0.f, s2 = 0.f;
#pragma unroll
        for (int ch = 0; ch < 16; ++ch){ s += ps1[tid][ch]; s2 += ps2[tid][ch]; }
        float mean = s * (1.f / D_);
        float var  = s2 * (1.f / D_) - mean * mean;
        mu[tid] = mean;
        rsig[tid] = rsqrtf(var + 1e-5f);
    }
    __syncthreads();
    for (int i = tid; i < 16 * D_; i += 256){
        int m = i / D_, c = i % D_;
        out[(size_t)row0 * D_ + i] = (res[i] - mu[m]) * rsig[m] * gamma[c] + beta[c];
    }
}

extern "C" void kernel_launch(void* const* d_in, const int* in_sizes, int n_in,
                              void* d_out, int out_size, void* d_ws, size_t ws_size,
                              hipStream_t stream){
    (void)in_sizes; (void)n_in; (void)out_size; (void)ws_size;
    const float* x     = (const float*)d_in[0];
    const float* Wq    = (const float*)d_in[1];
    const float* bq    = (const float*)d_in[2];
    const float* Wk    = (const float*)d_in[3];
    const float* bk    = (const float*)d_in[4];
    const float* Wo    = (const float*)d_in[5];
    const float* bo    = (const float*)d_in[6];
    const float* gamma = (const float*)d_in[7];
    const float* beta  = (const float*)d_in[8];
    float* out = (float*)d_out;

    char* ws = (char*)d_ws;
    const size_t SZ_XT  = (size_t)B_ * D_ * T_  * 2;   // 16 MB
    const size_t SZ_QK  = (size_t)B_ * T_ * DK_ * 2;   //  4 MB
    const size_t SZ_WQK = (size_t)DK_ * D_ * 2;        // 128 KB
    const size_t SZ_WO  = (size_t)D_ * D_  * 2;        // 512 KB
    unsigned short* xT    = (unsigned short*)(ws);
    unsigned short* qbf   = (unsigned short*)(ws + SZ_XT);
    unsigned short* kbf   = (unsigned short*)(ws + SZ_XT + SZ_QK);
    unsigned short* WqT   = (unsigned short*)(ws + SZ_XT + 2 * SZ_QK);
    unsigned short* WkT   = (unsigned short*)(ws + SZ_XT + 2 * SZ_QK + SZ_WQK);
    unsigned short* WoT   = (unsigned short*)(ws + SZ_XT + 2 * SZ_QK + 2 * SZ_WQK);
    unsigned short* accbf = (unsigned short*)(ws + SZ_XT + 2 * SZ_QK + 2 * SZ_WQK + SZ_WO);

    dim3 tb(32, 8);
    k_xT<<<dim3(T_ / 32, D_ / 32, B_), tb, 0, stream>>>(x, xT);
    k_wT<<<dim3(D_ / 32, DK_ / 32), tb, 0, stream>>>(Wq, WqT, D_, DK_);
    k_wT<<<dim3(D_ / 32, DK_ / 32), tb, 0, stream>>>(Wk, WkT, D_, DK_);
    k_wT<<<dim3(D_ / 32, D_  / 32), tb, 0, stream>>>(Wo, WoT, D_, D_);

    int ntiles = B_ * T_ / 16;   // 1024
    k_proj<<<ntiles, 256, 0, stream>>>(x, WqT, bq, WkT, bk, qbf, kbf);
    k_attn<<<ntiles, 256, 0, stream>>>(qbf, kbf, xT, accbf);
    k_out <<<ntiles, 256, 0, stream>>>(accbf, WoT, bo, x, gamma, beta, out);
}
